// VisualAttentionProcess_60181081752095
// MI455X (gfx1250) — compile-verified, matrix operand order checked by reading, not run on hardware
//
#include <hip/hip_runtime.h>
#include <hip/hip_bf16.h>

// ---------------------------------------------------------------------------
// MI455X / gfx1250: cross-attention + CORA value erase.
// Heavy GEMMs via v_wmma_f32_16x16x32_bf16 (wave32). q, k, v_new and attn
// intermediates kept in bf16: halves HBM round-trips and removes all f32->bf16
// conversion VALU from the attention kernel's fragment fills.
// ---------------------------------------------------------------------------

typedef __attribute__((ext_vector_type(16))) __bf16 v16bf;
typedef __attribute__((ext_vector_type(8)))  float  v8f;

#define CB   8
#define CN   4096
#define CC   640
#define CS   77
#define CXD  768
#define CH   8
#define CDH  80
#define CBETA 0.5f
#define CTAU  0.1f
#define CEPS  1e-8f

// Hardware bf16 conversion (gfx1250 is bf16-native; clang lowers this to the
// ISA convert instead of a 4-op software RNE sequence).
__device__ __forceinline__ __bf16 f2bf(float f) { return (__bf16)f; }
__device__ __forceinline__ __bf16 to_bf16(float x)  { return (__bf16)x; }
__device__ __forceinline__ __bf16 to_bf16(__bf16 x) { return x; }
__device__ __forceinline__ void   store_val(float*  p, float v) { *p = v; }
__device__ __forceinline__ void   store_val(__bf16* p, float v) { *p = (__bf16)v; }

// K-index inside a 16-bit A/B WMMA fragment (ISA 7.12.2 packing).
// Identity: frag_k(e,half) = (e<8) ? half*8+e : 16+half*8+(e-8)
__device__ __forceinline__ int frag_k(int e, int half) {
    return (e < 8) ? (half * 8 + e) : (16 + half * 8 + (e - 8));
}

// Branchless guarded row loads: always-in-bounds load + value select.
__device__ __forceinline__ __bf16 ld_s_clamped(const __bf16* __restrict__ base,
                                               int s, int d) {
    int sc = (s < CS) ? s : (CS - 1);
    __bf16 v = base[(long)sc * CC + d];
    return (s < CS) ? v : (__bf16)0.0f;
}

// ---------------------------------------------------------------------------
// Generic GEMM: C[M,Nc] = A[M,Kd] @ W[Kd,Nc] (+ bias). A: f32 or bf16 in,
// C: f32 or bf16 out. 256 threads = 8 waves; block tile 128x64; BK = 32.
// Each wave: 2 M-tiles x 2 N-tiles -> 4 WMMAs per k-step.
// ---------------------------------------------------------------------------
template <typename TA, typename TO>
__global__ __launch_bounds__(256)
void gemm_wmma(const TA* __restrict__ A, const float* __restrict__ W,
               const float* __restrict__ bias, TO* __restrict__ Cc,
               int M, int Kd, int Nc)
{
    __shared__ __bf16 sA[128 * 32];  // [m][k]
    __shared__ __bf16 sW[64 * 32];   // [n][k] (transposed: K-pairs contiguous)

    const int tid  = threadIdx.x;
    const int lane = tid & 31;
    const int wave = tid >> 5;
    const int r    = lane & 15;
    const int half = lane >> 4;

    const int n0blk = blockIdx.x * 64;
    const int m0blk = blockIdx.y * 128;
    const int m0    = (wave >> 1) * 32;   // two 16-row tiles per wave
    const int nb    = (wave & 1) * 32;    // two 16-col tiles per wave

    v8f acc00 = {}, acc01 = {}, acc10 = {}, acc11 = {};

    for (int k0 = 0; k0 < Kd; k0 += 32) {
        // stage A block [128 x 32] (coalesced along k); clamp+select guard
        for (int i = tid; i < 128 * 32; i += 256) {
            int mm = i >> 5, kk = i & 31;
            int gm = m0blk + mm;
            int gmc = (gm < M) ? gm : (M - 1);
            __bf16 v = to_bf16(A[(long)gmc * Kd + (k0 + kk)]);
            sA[i] = (gm < M) ? v : (__bf16)0.0f;
        }
        // stage W block [32 x 64] transposed (coalesced along n)
        for (int i = tid; i < 64 * 32; i += 256) {
            int kk = i >> 6, nn = i & 63;
            sW[nn * 32 + kk] = f2bf(W[(long)(k0 + kk) * Nc + n0blk + nn]);
        }
        // prefetch next k-step tiles (global_prefetch_b8)
        if (k0 + 32 < Kd) {
            int gp = m0blk + (tid >> 1);
            int gpc = (gp < M) ? gp : (M - 1);
            __builtin_prefetch(&A[(long)gpc * Kd + k0 + 32], 0, 1);
            __builtin_prefetch(&W[(long)(k0 + 32 + (tid >> 3)) * Nc + n0blk], 0, 1);
        }
        __syncthreads();

        v16bf af0, af1, b0, b1;
        #pragma unroll
        for (int e = 0; e < 16; ++e) {
            int kk = frag_k(e, half);
            af0[e] = sA[(m0 + r) * 32 + kk];
            af1[e] = sA[(m0 + 16 + r) * 32 + kk];
            b0[e]  = sW[(nb + r) * 32 + kk];
            b1[e]  = sW[(nb + 16 + r) * 32 + kk];
        }
        acc00 = __builtin_amdgcn_wmma_f32_16x16x32_bf16(false, af0, false, b0,
                                                        (short)0, acc00, false, false);
        acc01 = __builtin_amdgcn_wmma_f32_16x16x32_bf16(false, af0, false, b1,
                                                        (short)0, acc01, false, false);
        acc10 = __builtin_amdgcn_wmma_f32_16x16x32_bf16(false, af1, false, b0,
                                                        (short)0, acc10, false, false);
        acc11 = __builtin_amdgcn_wmma_f32_16x16x32_bf16(false, af1, false, b1,
                                                        (short)0, acc11, false, false);
        __syncthreads();
    }

    const int n0 = n0blk + nb + r;
    const int n1 = n0 + 16;
    if (bias) {
        const float bv0 = bias[n0], bv1 = bias[n1];
        #pragma unroll
        for (int i = 0; i < 8; ++i) {
            int ma = m0blk + m0 + i + half * 8;
            int mb = ma + 16;
            if (ma < M) {
                store_val(&Cc[(long)ma * Nc + n0], acc00[i] + bv0);
                store_val(&Cc[(long)ma * Nc + n1], acc01[i] + bv1);
            }
            if (mb < M) {
                store_val(&Cc[(long)mb * Nc + n0], acc10[i] + bv0);
                store_val(&Cc[(long)mb * Nc + n1], acc11[i] + bv1);
            }
        }
    } else {
        #pragma unroll
        for (int i = 0; i < 8; ++i) {
            int ma = m0blk + m0 + i + half * 8;
            int mb = ma + 16;
            if (ma < M) {
                store_val(&Cc[(long)ma * Nc + n0], acc00[i]);
                store_val(&Cc[(long)ma * Nc + n1], acc01[i]);
            }
            if (mb < M) {
                store_val(&Cc[(long)mb * Nc + n0], acc10[i]);
                store_val(&Cc[(long)mb * Nc + n1], acc11[i]);
            }
        }
    }
}

// ---------------------------------------------------------------------------
// CORA erase: one block per token s. Reads f32 v, writes bf16 v_new.
// ---------------------------------------------------------------------------
__device__ __forceinline__ float block_sum(float v, float* red) {
    int tid = threadIdx.x;
    red[tid] = v; __syncthreads();
    for (int off = 128; off > 0; off >>= 1) {
        if (tid < off) red[tid] += red[tid + off];
        __syncthreads();
    }
    float r = red[0]; __syncthreads();
    return r;
}

__global__ __launch_bounds__(256)
void cora_kernel(const float* __restrict__ v, const float* __restrict__ tg,
                 const float* __restrict__ an, const float* __restrict__ pr,
                 __bf16* __restrict__ vn)
{
    __shared__ float red[256];
    const int s = blockIdx.x, tid = threadIdx.x;
    const float* p1 = pr + (long)(0 * CS + s) * CC;
    const float* p2 = pr + (long)(1 * CS + s) * CC;
    const float* u  = tg + (long)s * CC;
    const float* a  = an + (long)s * CC;

    float l11 = 0, l12 = 0, l22 = 0, lu1 = 0, lu2 = 0, la1 = 0, la2 = 0;
    for (int d = tid; d < CC; d += 256) {
        float x1 = p1[d], x2 = p2[d], uu = u[d], aa = a[d];
        l11 += x1 * x1; l12 += x1 * x2; l22 += x2 * x2;
        lu1 += x1 * uu; lu2 += x2 * uu; la1 += x1 * aa; la2 += x2 * aa;
    }
    float g11 = block_sum(l11, red), g12 = block_sum(l12, red), g22 = block_sum(l22, red);
    float bu1 = block_sum(lu1, red), bu2 = block_sum(lu2, red);
    float ba1 = block_sum(la1, red), ba2 = block_sum(la2, red);
    float invdet = 1.0f / (g11 * g22 - g12 * g12);
    float cu1 = ( g22 * bu1 - g12 * bu2) * invdet;
    float cu2 = (-g12 * bu1 + g11 * bu2) * invdet;
    float ca1 = ( g22 * ba1 - g12 * ba2) * invdet;
    float ca2 = (-g12 * ba1 + g11 * ba2) * invdet;

    float suu = 0, sua = 0;
    for (int d = tid; d < CC; d += 256) {
        float ud = u[d] - cu1 * p1[d] - cu2 * p2[d];
        float ad = a[d] - ca1 * p1[d] - ca2 * p2[d];
        suu += ud * ud; sua += ud * ad;
    }
    suu = block_sum(suu, red); sua = block_sum(sua, red);
    float inv_nu = 1.0f / (sqrtf(suu) + CEPS);
    float ta = sua * inv_nu;                 // <u_hat, a_def>

    float sat = 0;
    for (int d = tid; d < CC; d += 256) {
        float ud = u[d] - cu1 * p1[d] - cu2 * p2[d];
        float ad = a[d] - ca1 * p1[d] - ca2 * p2[d];
        float at = ad - ta * (ud * inv_nu);
        sat += at * at;
    }
    sat = block_sum(sat, red);
    float inv_na = 1.0f / (sqrtf(sat) + CEPS);

    for (int bb = 0; bb < CB; ++bb) {
        const float* x = v + ((long)bb * CS + s) * CC;
        float b1 = 0, b2 = 0;
        for (int d = tid; d < CC; d += 256) { b1 += p1[d] * x[d]; b2 += p2[d] * x[d]; }
        b1 = block_sum(b1, red); b2 = block_sum(b2, red);
        float cx1 = ( g22 * b1 - g12 * b2) * invdet;
        float cx2 = (-g12 * b1 + g11 * b2) * invdet;

        float lt = 0, lnf = 0;
        for (int d = tid; d < CC; d += 256) {
            float vf = x[d] - cx1 * p1[d] - cx2 * p2[d];
            float uh = (u[d] - cu1 * p1[d] - cu2 * p2[d]) * inv_nu;
            lt += uh * vf; lnf += vf * vf;
        }
        float t  = block_sum(lt, red);
        float nf = block_sum(lnf, red);
        float denom = sqrtf(nf) + CEPS;
        int keep = (fabsf(t) / denom) < CTAU;

        __bf16* o = vn + ((long)bb * CS + s) * CC;
        for (int d = tid; d < CC; d += 256) {
            float uh = (u[d] - cu1 * p1[d] - cu2 * p2[d]) * inv_nu;
            float ad = a[d] - ca1 * p1[d] - ca2 * p2[d];
            float ah = (ad - ta * uh) * inv_na;
            float vo = keep ? x[d] : (x[d] - t * uh + CBETA * t * ah);
            o[d] = f2bf(vo);
        }
    }
}

// ---------------------------------------------------------------------------
// Attention: block = (batch b, 16 rows of N); 8 waves = 8 heads.
// q/K/V all bf16 -> fragment fills are pure loads (no conversions). K-chunk
// loops fully unrolled (guards fold statically); S-dim guards are branchless
// clamp+select. Zero barriers; per-wave LDS round-trip re-lays P.
// ---------------------------------------------------------------------------
__global__ __launch_bounds__(256)
void attn_kernel(const __bf16* __restrict__ Q, const __bf16* __restrict__ Kv,
                 const __bf16* __restrict__ Vv, __bf16* __restrict__ attn)
{
    __shared__ __bf16 P_lds[CH * 16 * CDH];   // per-head probs, [h][m][s]

    const int tid  = threadIdx.x;
    const int lane = tid & 31;
    const int h    = tid >> 5;                // wave id == head id
    const int r    = lane & 15;
    const int half = lane >> 4;

    const int ntiles = CN / 16;
    const int b  = blockIdx.x / ntiles;
    const int n0 = (blockIdx.x % ntiles) * 16;
    const float scale = 0.1118033988749895f;  // 1/sqrt(80)

    const __bf16* Qb = Q  + ((long)b * CN + n0) * CC + h * CDH;
    const __bf16* Kb = Kv + (long)b * CS * CC + h * CDH;
    const __bf16* Vb = Vv + (long)b * CS * CC + h * CDH;

    // ---- scores[16,80] = q_h @ K_h^T ----
    v8f sc[5] = {v8f{}, v8f{}, v8f{}, v8f{}, v8f{}};
    #pragma unroll
    for (int kc = 0; kc < CDH; kc += 32) {    // kc in {0,32,64}, compile-time
        v16bf af, bv;
        #pragma unroll
        for (int e = 0; e < 16; ++e) {
            int d = kc + frag_k(e, half);
            af[e] = (d < CDH) ? Qb[(long)r * CC + d] : (__bf16)0.0f;  // folds statically
        }
        #pragma unroll
        for (int st = 0; st < 5; ++st) {
            int s = st * 16 + r;              // divergent only for st==4
            #pragma unroll
            for (int e = 0; e < 16; ++e) {
                int d = kc + frag_k(e, half);
                bv[e] = (d < CDH) ? ld_s_clamped(Kb, s, d) : (__bf16)0.0f;
            }
            sc[st] = __builtin_amdgcn_wmma_f32_16x16x32_bf16(false, af, false, bv,
                                                             (short)0, sc[st], false, false);
        }
    }

    // ---- softmax over s (rows live in 16-lane halves) ----
    #pragma unroll
    for (int st = 0; st < 5; ++st) {
        int s = st * 16 + r;
        #pragma unroll
        for (int i = 0; i < 8; ++i)
            sc[st][i] = (s < CS) ? sc[st][i] * scale : -1e30f;
    }
    #pragma unroll
    for (int i = 0; i < 8; ++i) {
        float mx = -1e30f;
        #pragma unroll
        for (int st = 0; st < 5; ++st) mx = fmaxf(mx, sc[st][i]);
        #pragma unroll
        for (int off = 8; off >= 1; off >>= 1) mx = fmaxf(mx, __shfl_xor(mx, off, 32));
        float sum = 0.0f;
        #pragma unroll
        for (int st = 0; st < 5; ++st) {
            float ev = __expf(sc[st][i] - mx);
            sc[st][i] = ev; sum += ev;
        }
        #pragma unroll
        for (int off = 8; off >= 1; off >>= 1) sum += __shfl_xor(sum, off, 32);
        float inv = 1.0f / sum;
        #pragma unroll
        for (int st = 0; st < 5; ++st) sc[st][i] *= inv;
    }

    // re-lay probs (C-layout -> A-fragment layout) via per-wave LDS
    __bf16* P = P_lds + h * (16 * CDH);
    #pragma unroll
    for (int st = 0; st < 5; ++st)
        #pragma unroll
        for (int i = 0; i < 8; ++i) {
            int m = i + half * 8, s = st * 16 + r;
            P[m * CDH + s] = f2bf((s < CS) ? sc[st][i] : 0.0f);
        }

    // ---- out[16,80] = P @ V_h ----
    v8f ov[5] = {v8f{}, v8f{}, v8f{}, v8f{}, v8f{}};
    #pragma unroll
    for (int kc = 0; kc < CDH; kc += 32) {    // kc in {0,32,64}, compile-time
        v16bf af, bv;
        #pragma unroll
        for (int e = 0; e < 16; ++e) {
            int s = kc + frag_k(e, half);
            af[e] = (s < CDH) ? P[r * CDH + s] : (__bf16)0.0f;      // folds statically
        }
        #pragma unroll
        for (int dt = 0; dt < 5; ++dt) {
            int d = dt * 16 + r;
            #pragma unroll
            for (int e = 0; e < 16; ++e) {
                int s = kc + frag_k(e, half);
                bv[e] = (s < CDH) ? ld_s_clamped(Vb, s, d) : (__bf16)0.0f;
            }
            ov[dt] = __builtin_amdgcn_wmma_f32_16x16x32_bf16(false, af, false, bv,
                                                             (short)0, ov[dt], false, false);
        }
    }

    #pragma unroll
    for (int dt = 0; dt < 5; ++dt)
        #pragma unroll
        for (int i = 0; i < 8; ++i) {
            int m = i + half * 8;
            int d = dt * 16 + r;
            attn[((long)b * CN + n0 + m) * CC + h * CDH + d] = f2bf(ov[dt][i]);
        }
}

// ---------------------------------------------------------------------------
// Launcher
// ---------------------------------------------------------------------------
extern "C" void kernel_launch(void* const* d_in, const int* in_sizes, int n_in,
                              void* d_out, int out_size, void* d_ws, size_t ws_size,
                              hipStream_t stream) {
    const float* hs  = (const float*)d_in[0];
    const float* enc = (const float*)d_in[1];
    const float* Wq  = (const float*)d_in[2];
    const float* Wk  = (const float*)d_in[3];
    const float* Wv  = (const float*)d_in[4];
    const float* Wo  = (const float*)d_in[5];
    const float* bo  = (const float*)d_in[6];
    const float* tg  = (const float*)d_in[7];
    const float* an  = (const float*)d_in[8];
    const float* pr  = (const float*)d_in[9];
    float* out = (float*)d_out;

    const long kv_elems = (long)CB * CS * CC;         // 616*640
    const long qn_elems = (long)CB * CN * CC;         // 32768*640
    float*  v_ws    = (float*)d_ws;                   // f32 first (alignment)
    __bf16* k_ws    = (__bf16*)(v_ws + kv_elems);     // bf16 [B*S, C]
    __bf16* vn_ws   = k_ws + kv_elems;                // bf16 [B*S, C]
    __bf16* q_ws    = vn_ws + kv_elems;               // bf16 [B*N, C]
    __bf16* attn_ws = q_ws + qn_elems;                // bf16 [B*N, C]

    dim3 blk(256);
    const int Mkv = CB * CS;                          // 616
    const int Mq  = CB * CN;                          // 32768

    // q = HS @ Wq   (f32 in -> bf16 out)
    dim3 g_q(CC / 64, (Mq + 127) / 128);
    gemm_wmma<float, __bf16><<<g_q, blk, 0, stream>>>(hs, Wq, nullptr, q_ws, Mq, CC, CC);

    // k = E @ Wk (bf16 out) ; v = E @ Wv (f32 out, CORA consumes full precision)
    dim3 g_kv(CC / 64, (Mkv + 127) / 128);
    gemm_wmma<float, __bf16><<<g_kv, blk, 0, stream>>>(enc, Wk, nullptr, k_ws, Mkv, CXD, CC);
    gemm_wmma<float, float ><<<g_kv, blk, 0, stream>>>(enc, Wv, nullptr, v_ws, Mkv, CXD, CC);

    // CORA erase on v (f32 in -> bf16 out)
    cora_kernel<<<dim3(CS), blk, 0, stream>>>(v_ws, tg, an, pr, vn_ws);

    // attention (q/K/V bf16 -> attn bf16)
    attn_kernel<<<dim3(CB * (CN / 16)), blk, 0, stream>>>(q_ws, k_ws, vn_ws, attn_ws);

    // out = attn @ Wo + bo  (bf16 in -> f32 out)
    gemm_wmma<__bf16, float><<<g_q, blk, 0, stream>>>(attn_ws, Wo, bo, out, Mq, CC, CC);
}
